// PyramidalNeuron_80719615361696
// MI455X (gfx1250) — matrix operand order repaired and checked by reading, not compiled, not run on hardware
//
#include <hip/hip_runtime.h>

#define BATCH   4096
#define IMG     784
#define BASAL   16384
#define NCLS    10
#define KACT    492
#define KPAD    800      // 784 padded to 25 * 32
#define KSTEPS  25
#define LDS_STRIDE 40    // elements per LDS tile row (80B: 16B-aligned, conflict-free 20-bank stride)

typedef __attribute__((ext_vector_type(16))) __bf16 v16bf;
typedef __attribute__((ext_vector_type(8)))  float  v8f;

// ---------- helpers ----------

__device__ __forceinline__ unsigned short f2bf(float f) {
  unsigned u = __float_as_uint(f);
  u += 0x7FFFu + ((u >> 16) & 1u);      // round-to-nearest-even
  return (unsigned short)(u >> 16);
}

__device__ __forceinline__ unsigned sortable(float f) {
  unsigned u = __float_as_uint(f);
  return u ^ ((u >> 31) ? 0xFFFFFFFFu : 0x80000000u);
}

__device__ __forceinline__ float unsortable(unsigned u) {
  u = (u >> 31) ? (u ^ 0x80000000u) : ~u;
  return __uint_as_float(u);
}

union FragU { uint4 u[2]; v16bf v; };

// A 16x32 bf16 fragment: lanes 0-15 hold K {0-7,16-23}, lanes 16-31 hold K {8-15,24-31}
__device__ __forceinline__ v16bf load_a(const unsigned short* p, int hi) {
  FragU f;
  f.u[0] = *(const uint4*)(p + (hi ? 8 : 0));
  f.u[1] = *(const uint4*)(p + (hi ? 24 : 16));
  return f.v;
}
// B 32x16 bf16 fragment (N-major rows): lanes 0-15 hold K {0-7,16-23}, lanes 16-31 hold K {8-15,24-31}
__device__ __forceinline__ v16bf load_b(const unsigned short* p, int hi) {
  FragU f;
  f.u[0] = *(const uint4*)(p + (hi ? 8 : 0));
  f.u[1] = *(const uint4*)(p + (hi ? 24 : 16));
  return f.v;
}

// Async-copy a 128-row x 32-element bf16 tile (row stride KPAD in global) into LDS
// (row stride LDS_STRIDE). 512 x 16B chunks, 2 chunks per thread, 2 async ops/wave-instr
// per thread -> 4 async instructions per wave per (A,B) tile pair.
__device__ __forceinline__ void async_tile_load(const unsigned short* gbase,
                                                unsigned ldsBase, int tid) {
  #pragma unroll
  for (int it = 0; it < 2; ++it) {
    int i   = tid + it * 256;
    int row = i >> 2, q = i & 3;
    unsigned long long g = (unsigned long long)(uintptr_t)(gbase + (size_t)row * KPAD + q * 8);
    unsigned l = ldsBase + (unsigned)(row * (LDS_STRIDE * 2) + q * 16);
    asm volatile("global_load_async_to_lds_b128 %0, %1, off"
                 :: "v"(l), "v"(g) : "memory");
  }
}

// ---------- kernel 1: image f32 -> bf16, K padded to 800 ----------
__global__ void cvt_image(const float* __restrict__ img, unsigned short* __restrict__ out) {
  int idx = blockIdx.x * blockDim.x + threadIdx.x;
  if (idx >= BATCH * KPAD) return;
  int r = idx / KPAD, k = idx - r * KPAD;
  float v = (k < IMG) ? img[r * IMG + k] : 0.0f;
  out[idx] = f2bf(v);
}

// ---------- kernel 2: projection [784,16384] f32 -> transposed bf16 [16384,800] ----------
__global__ void cvt_projT(const float* __restrict__ proj, unsigned short* __restrict__ out) {
  __shared__ float tile[32][33];
  int kt = blockIdx.x * 32;
  int nt = blockIdx.y * 32;
  int tx = threadIdx.x, ty = threadIdx.y;   // (32, 8)
  #pragma unroll
  for (int i = 0; i < 32; i += 8) {
    int k = kt + ty + i;
    tile[ty + i][tx] = (k < IMG) ? proj[k * BASAL + nt + tx] : 0.0f;
  }
  __syncthreads();
  #pragma unroll
  for (int i = 0; i < 32; i += 8) {
    int n = nt + ty + i;
    out[n * KPAD + kt + tx] = f2bf(tile[tx][ty + i]);
  }
}

// ---------- kernel 3: WMMA bf16 GEMM with async-LDS double buffering ----------
// block = 256 threads = 8 waves, block tile 128M x 128N.
// wave (waveM, waveN) computes 32M x 64N: 2 A-frags x 4 B-frags -> 8 WMMA per K-step.
__global__ void __launch_bounds__(256) gemm_wmma(const unsigned short* __restrict__ A,
                                                 const unsigned short* __restrict__ Bt,
                                                 float* __restrict__ C) {
  __shared__ unsigned short ldsA[2][128 * LDS_STRIDE];
  __shared__ unsigned short ldsB[2][128 * LDS_STRIDE];

  int tid   = threadIdx.x;
  int lane  = tid & 31;
  int wave  = tid >> 5;
  int lhalf = lane >> 4;
  int l16   = lane & 15;
  int waveM = wave >> 1;          // 0..3 -> 32-row strip
  int waveN = wave & 1;           // 0..1 -> 64-col strip
  int m0    = blockIdx.y * 128;
  int n0    = blockIdx.x * 128;

  const unsigned short* aTile = A  + (size_t)m0 * KPAD;
  const unsigned short* bTile = Bt + (size_t)n0 * KPAD;

  unsigned la0 = (unsigned)(uintptr_t)&ldsA[0][0];
  unsigned la1 = (unsigned)(uintptr_t)&ldsA[1][0];
  unsigned lb0 = (unsigned)(uintptr_t)&ldsB[0][0];
  unsigned lb1 = (unsigned)(uintptr_t)&ldsB[1][0];

  // prefetch tile 0 into buffer 0
  async_tile_load(aTile, la0, tid);
  async_tile_load(bTile, lb0, tid);

  v8f acc[2][4] = {};

  for (int kt = 0; kt < KSTEPS; ++kt) {
    int cur = kt & 1;
    if (kt + 1 < KSTEPS) {
      const unsigned short* aN = aTile + (kt + 1) * 32;
      const unsigned short* bN = bTile + (kt + 1) * 32;
      async_tile_load(aN, cur ? la0 : la1, tid);
      async_tile_load(bN, cur ? lb0 : lb1, tid);
      asm volatile("s_wait_asynccnt 0x4" ::: "memory");   // tile kt landed
    } else {
      asm volatile("s_wait_asynccnt 0x0" ::: "memory");
    }
    __syncthreads();

    const unsigned short* la = &ldsA[cur][0];
    const unsigned short* lb = &ldsB[cur][0];
    v16bf af[2], bfr[4];
    #pragma unroll
    for (int mi = 0; mi < 2; ++mi)
      af[mi] = load_a(la + (waveM * 32 + mi * 16 + l16) * LDS_STRIDE, lhalf);
    #pragma unroll
    for (int ni = 0; ni < 4; ++ni)
      bfr[ni] = load_b(lb + (waveN * 64 + ni * 16 + l16) * LDS_STRIDE, lhalf);
    #pragma unroll
    for (int mi = 0; mi < 2; ++mi) {
      #pragma unroll
      for (int ni = 0; ni < 4; ++ni) {
        acc[mi][ni] = __builtin_amdgcn_wmma_f32_16x16x32_bf16(
            false, af[mi], false, bfr[ni], (short)0, acc[mi][ni], false, false);
      }
    }
    __syncthreads();   // all waves done reading buf[cur] before it is refilled
  }

  // C/D layout: lanes 0-15 -> M = r, lanes 16-31 -> M = 8 + r; N = lane%16
  #pragma unroll
  for (int mi = 0; mi < 2; ++mi) {
    int rowBase = m0 + waveM * 32 + mi * 16 + (lhalf ? 8 : 0);
    #pragma unroll
    for (int r = 0; r < 8; ++r) {
      float* cr = C + (size_t)(rowBase + r) * BASAL + n0 + waveN * 64 + l16;
      #pragma unroll
      for (int ni = 0; ni < 4; ++ni)
        cr[ni * 16] = acc[mi][ni][r];
    }
  }
}

// ---------- kernel 4: per-row radix select of 492nd-largest value ----------
__global__ void __launch_bounds__(256) select_k(const float* __restrict__ act,
                                                float* __restrict__ thr,
                                                int* __restrict__ needOut) {
  int row = blockIdx.x;
  const float* a = act + (size_t)row * BASAL;
  __shared__ unsigned hist[256];
  __shared__ unsigned s_prefix, s_remaining;
  int tid = threadIdx.x;
  if (tid == 0) { s_prefix = 0; s_remaining = KACT; }
  __syncthreads();

  for (int pass = 0; pass < 4; ++pass) {
    int shift = 24 - 8 * pass;
    hist[tid] = 0;
    __syncthreads();
    unsigned pfx = s_prefix;
    for (int j = tid; j < BASAL; j += 256) {
      unsigned key = sortable(a[j]);
      if (pass == 0 || (key >> (shift + 8)) == pfx)
        atomicAdd(&hist[(key >> shift) & 255u], 1u);
    }
    __syncthreads();
    if (tid == 0) {
      unsigned rem = s_remaining, cum = 0;
      int b = 255;
      for (; b > 0; --b) {
        if (cum + hist[b] >= rem) break;
        cum += hist[b];
      }
      s_remaining = rem - cum;                  // ties still needed
      s_prefix = (s_prefix << 8) | (unsigned)b;
    }
    __syncthreads();
  }
  if (tid == 0) {
    thr[row] = unsortable(s_prefix);
    needOut[row] = (int)s_remaining;
  }
}

// ---------- kernel 5: masked overlap with index-ordered tie breaking ----------
__global__ void __launch_bounds__(256) overlap_k(const float* __restrict__ act,
                                                 const float* __restrict__ syn,
                                                 const float* __restrict__ thr,
                                                 const int* __restrict__ needArr,
                                                 float* __restrict__ out) {
  int row = blockIdx.x;
  const float* a = act + (size_t)row * BASAL;
  float T = thr[row];
  unsigned need = (unsigned)needArr[row];
  __shared__ unsigned eqcnt[256];
  __shared__ float accs[NCLS];
  int tid = threadIdx.x;
  int j0 = tid * 64;

  unsigned cnt = 0;
  for (int j = j0; j < j0 + 64; ++j) cnt += (a[j] == T);
  eqcnt[tid] = cnt;
  if (tid < NCLS) accs[tid] = 0.0f;
  __syncthreads();
  if (tid == 0) {
    unsigned run = 0;
    for (int i = 0; i < 256; ++i) { unsigned c = eqcnt[i]; eqcnt[i] = run; run += c; }
  }
  __syncthreads();

  unsigned rank = eqcnt[tid];
  float acc[NCLS] = {};
  for (int j = j0; j < j0 + 64; ++j) {
    float v = a[j];
    bool take = v > T;
    if (v == T) { if (rank < need) take = true; rank++; }
    if (take) {
      #pragma unroll
      for (int c = 0; c < NCLS; ++c) acc[c] += syn[c * BASAL + j];
    }
  }
  #pragma unroll
  for (int c = 0; c < NCLS; ++c) atomicAdd(&accs[c], acc[c]);
  __syncthreads();
  if (tid < NCLS) out[row * NCLS + tid] = accs[tid];
}

// ---------- launcher ----------
extern "C" void kernel_launch(void* const* d_in, const int* in_sizes, int n_in,
                              void* d_out, int out_size, void* d_ws, size_t ws_size,
                              hipStream_t stream) {
  const float* image = (const float*)d_in[0];   // [4096, 784]
  const float* proj  = (const float*)d_in[1];   // [784, 16384]
  const float* syn   = (const float*)d_in[2];   // [10, 16384]
  float* out = (float*)d_out;                   // [4096, 10]

  char* w = (char*)d_ws;
  size_t offImg  = 0;
  size_t offProj = offImg  + (size_t)BATCH * KPAD * 2;
  size_t offAct  = offProj + (size_t)BASAL * KPAD * 2;
  size_t offThr  = offAct  + (size_t)BATCH * BASAL * 4;
  size_t offNeed = offThr  + (size_t)BATCH * 4;

  unsigned short* imgB  = (unsigned short*)(w + offImg);
  unsigned short* projB = (unsigned short*)(w + offProj);
  float*          act   = (float*)(w + offAct);
  float*          thr   = (float*)(w + offThr);
  int*            need  = (int*)(w + offNeed);

  cvt_image<<<(BATCH * KPAD + 255) / 256, 256, 0, stream>>>(image, imgB);
  cvt_projT<<<dim3(KPAD / 32, BASAL / 32), dim3(32, 8), 0, stream>>>(proj, projB);
  gemm_wmma<<<dim3(BASAL / 128, BATCH / 128), 256, 0, stream>>>(imgB, projB, act);
  select_k<<<BATCH, 256, 0, stream>>>(act, thr, need);
  overlap_k<<<BATCH, 256, 0, stream>>>(act, syn, thr, need, out);
}
